// MHSelfAttention_67379446940107
// MI455X (gfx1250) — matrix operand order repaired and checked by reading, not compiled, not run on hardware
//
#include <hip/hip_runtime.h>
#include <hip/hip_bf16.h>

// MHA prefill for B=1, S=2048, E=2048, H=16, HD=128 on gfx1250 (wave32).
// GEMMs via v_wmma_f32_16x16x32_bf16 with double-buffered async LDS staging
// of the shared weight tile (global_load_async_to_lds_b128 / ASYNCcnt),
// software-pipelined 2 deep (A fragments) and 1 deep (B fragments from LDS).

typedef __bf16 bf16_t;
typedef __attribute__((ext_vector_type(16))) __bf16 v16bf;
typedef __attribute__((ext_vector_type(8)))  __bf16 v8bf;
typedef __attribute__((ext_vector_type(8)))  float  v8f;

#define S_LEN 2048
#define EMB   2048
#define NH    16
#define HD    128
#define NF    6144   // 3*EMB
#define WROW  40     // padded LDS row stride (bf16) for 32-element rows
#define BN    128    // block N-tile (rows of W staged per k-step)

__device__ __forceinline__ v8f wmma_bf16(v16bf a, v16bf b, v8f c) {
  return __builtin_amdgcn_wmma_f32_16x16x32_bf16(
      /*neg_a=*/false, a, /*neg_b=*/false, b,
      /*c_mod=*/(short)0, c, /*reuse_a=*/false, /*reuse_b=*/false);
}

// Combine two 16B-aligned 8-element chunks into one 16-element fragment.
__device__ __forceinline__ v16bf combine8(const bf16_t* lo, const bf16_t* hi) {
  v8bf a0 = *(const v8bf*)lo;
  v8bf a1 = *(const v8bf*)hi;
  v16bf r;
#pragma unroll
  for (int i = 0; i < 8; ++i) { r[i] = a0[i]; r[i + 8] = a1[i]; }
  return r;
}

// A-fragment (16x32 bf16): lane holds row m=lane&15; K values are two
// contiguous 8-element chunks: [h*8, h*8+8) and [16+h*8, 16+h*8+8) in the GEMMs; [16h, 16h+16) in attn.
__device__ __forceinline__ v16bf load_a_frag(const bf16_t* lo, const bf16_t* hi) {
  return combine8(lo, hi);
}

// LDS byte address of a __shared__ object (low 32 bits of generic pointer).
__device__ __forceinline__ unsigned lds_addr_of(const void* p) {
  return (unsigned)(unsigned long long)p;
}

// Async DMA: 16 bytes per lane, global -> LDS, tracked by ASYNCcnt.
__device__ __forceinline__ void async_copy_b128(unsigned lds, const void* gaddr) {
  asm volatile("global_load_async_to_lds_b128 %0, %1, off"
               :: "v"(lds), "v"(gaddr)
               : "memory");
}

__device__ __forceinline__ void wait_async0() {
  asm volatile("s_wait_asynccnt 0" ::: "memory");
}

// ---------------- f32 -> bf16 convert ----------------
__global__ void cvt_bf16_kernel(const float* __restrict__ src,
                                bf16_t* __restrict__ dst, int n) {
  int i = blockIdx.x * blockDim.x + threadIdx.x;
  int stride = gridDim.x * blockDim.x;
  for (; i < n; i += stride) dst[i] = (bf16_t)src[i];
}

// B fragment (32x16) from the LDS tile for n-subtile nt.
__device__ __forceinline__ v16bf ldb(const bf16_t* wl, int nt, int l16,
                                     int half) {
  const bf16_t* bp = wl + (nt * 16 + l16) * WROW + half * 8;
  return combine8(bp, bp + 16);
}

// One k-step of the wave's 16x128 tile: 8 WMMAs, B fragments pipelined one
// deep so each WMMA overlaps the ds_load pair of the next one.
__device__ __forceinline__ void compute_step(const bf16_t* wl, v16bf a,
                                             v8f acc[8], int l16, int half) {
  v16bf bcur = ldb(wl, 0, l16, half);
#pragma unroll
  for (int nt = 0; nt < 8; ++nt) {
    v16bf bnxt;
    if (nt < 7) bnxt = ldb(wl, nt + 1, l16, half);
    acc[nt] = wmma_bf16(a, bcur, acc[nt]);
    if (nt < 7) bcur = bnxt;
  }
}

// ---------------- block GEMM core with async LDS staging ----------------
// C[m][n] = sum_k A[m][k] * W[n][k].  A: MxK row-major, W: NxK row-major.
// Block = 8 waves sharing W rows [n0, n0+BN) staged per 32-k-step into a
// double-buffered LDS tile (global_load_async_to_lds_b128, 2 per thread).
// Software-pipelined 2 deep: a0/a1 alternate, buffers chosen statically.
__device__ __forceinline__ void gemm_block_16x128(const bf16_t* __restrict__ A,
                                                  int lda,
                                                  const bf16_t* __restrict__ W,
                                                  int ldw, int m0_wave, int n0,
                                                  int K, bf16_t* Wl,
                                                  v8f acc[8]) {
  const int tid  = threadIdx.x;
  const int lane = tid & 31;
  const int half = lane >> 4;
  const int l16  = lane & 15;
  // staging: BN rows x 64B = 512 x 16B chunks; 2 chunks per thread
  const int srow0 = tid >> 1;               // chunk 0 row (0..127)
  const int sseg0 = (tid & 1) * 2;          // chunk 0 seg  (0 or 2)
  const bf16_t* gsrc0 = W + (size_t)(n0 + srow0) * ldw + sseg0 * 8;
  const bf16_t* gsrc1 = gsrc0 + 8;          // chunk 1: next 16B segment
  const unsigned lc0  = lds_addr_of(Wl + srow0 * WROW + sseg0 * 8);
  const unsigned lc1  = lc0 + 16;
  const unsigned bufb = BN * WROW * (unsigned)sizeof(bf16_t);
  const bf16_t* arow = A + (size_t)(m0_wave + l16) * lda;

  // prologue: stage k-step 0 into buffer 0; load first A fragment
  async_copy_b128(lc0, gsrc0);
  async_copy_b128(lc1, gsrc1);
  v16bf a0 = load_a_frag(arow + half * 8, arow + 16 + half * 8);
  v16bf a1;

  const int nsteps = K / 32;  // even (K = 2048 -> 64)
#pragma unroll 1
  for (int i = 0; i < nsteps; i += 2) {
    // ======== even step: compute from buffer 0, stage buffer 1 ========
    wait_async0();
    __syncthreads();
    {
      const size_t kn = (size_t)(i + 1) * 32;
      async_copy_b128(bufb + lc0, gsrc0 + kn);
      async_copy_b128(bufb + lc1, gsrc1 + kn);
      a1 = load_a_frag(arow + kn + half * 8, arow + kn + 16 + half * 8);
      compute_step(Wl, a0, acc, l16, half);
    }
    // ======== odd step: compute from buffer 1, stage buffer 0 ========
    wait_async0();
    __syncthreads();
    {
      const size_t kn = (i + 2 < nsteps) ? (size_t)(i + 2) * 32
                                         : (size_t)i * 32;  // clamped reload
      if (i + 2 < nsteps) {
        async_copy_b128(lc0, gsrc0 + kn);
        async_copy_b128(lc1, gsrc1 + kn);
      }
      a0 = load_a_frag(arow + kn + half * 8, arow + kn + 16 + half * 8);
      compute_step(Wl + BN * WROW, a1, acc, l16, half);
    }
  }
}

// ---------------- QKV projection ----------------
// qkv[m][f] -> split f = hh*384 + c into q/k (row-major [s][h][hd], bf16)
// and v transposed vT[h][hd][s] so PV B-fragments are contiguous.
__global__ void gemm_qkv_kernel(const bf16_t* __restrict__ X,
                                const bf16_t* __restrict__ Wqkv,
                                bf16_t* __restrict__ q, bf16_t* __restrict__ k,
                                bf16_t* __restrict__ vT) {
  __shared__ bf16_t Wl[2 * BN * WROW];
  const int lane = threadIdx.x & 31;
  const int wv   = threadIdx.x >> 5;
  const int half = lane >> 4;
  const int l16  = lane & 15;
  const int ntn  = NF / BN;               // 48
  const int bn   = blockIdx.x % ntn;
  const int bm   = blockIdx.x / ntn;
  const int m0   = bm * 128 + wv * 16;
  const int n0   = bn * BN;

  v8f acc[8];
#pragma unroll
  for (int nt = 0; nt < 8; ++nt)
#pragma unroll
    for (int r = 0; r < 8; ++r) acc[nt][r] = 0.0f;

  gemm_block_16x128(X, EMB, Wqkv, EMB, m0, n0, EMB, Wl, acc);

#pragma unroll
  for (int nt = 0; nt < 8; ++nt) {
#pragma unroll
    for (int r = 0; r < 8; ++r) {
      int n = n0 + nt * 16 + l16;
      int m = m0 + half * 8 + r;
      bf16_t val = (bf16_t)acc[nt][r];
      int hh = n / 384;
      int c  = n % 384;
      if (c < 128)
        q[((size_t)m * NH + hh) * HD + c] = val;
      else if (c < 256)
        k[((size_t)m * NH + hh) * HD + (c - 128)] = val;
      else
        vT[((size_t)hh * HD + (c - 256)) * S_LEN + m] = val;
    }
  }
}

// ---------------- RoPE (in place on bf16 q or k) ----------------
__global__ void rope_kernel(bf16_t* __restrict__ t,
                            const float* __restrict__ cs,
                            const float* __restrict__ sn) {
  int idx = blockIdx.x * blockDim.x + threadIdx.x;  // over S*H*(HD/2)
  if (idx >= S_LEN * NH * (HD / 2)) return;
  int i  = idx % (HD / 2);
  int hh = (idx / (HD / 2)) % NH;
  int s  = idx / ((HD / 2) * NH);
  size_t base = ((size_t)s * NH + hh) * HD + 2 * i;
  float xr = (float)t[base], xi = (float)t[base + 1];
  float c = cs[s * (HD / 2) + i], si = sn[s * (HD / 2) + i];
  t[base]     = (bf16_t)(xr * c - xi * si);
  t[base + 1] = (bf16_t)(xr * si + xi * c);
}

// ---------------- flash attention ----------------
// 1 wave = 16 q-rows of one head; block = 8 waves (128 q-rows, same head).
__global__ void attn_kernel(const bf16_t* __restrict__ q,
                            const bf16_t* __restrict__ k,
                            const bf16_t* __restrict__ vT,
                            bf16_t* __restrict__ y) {
  __shared__ float sc[8][16][32];   // per-wave 16x32 score/prob tile
  __shared__ float sbc[8][16];      // per-wave row broadcast (alpha / l)
  const int lane = threadIdx.x & 31;
  const int wv   = threadIdx.x >> 5;
  const int half = lane >> 4;
  const int l16  = lane & 15;
  const int head = blockIdx.x >> 4;
  const int grp  = blockIdx.x & 15;
  const int qt   = (grp * 8 + wv) * 16;
  const float scale = 0.08838834764831845f;  // 1/sqrt(128)

  // Preload Q A-fragments for all 4 k-chunks of HD=128.
  const bf16_t* qrow = q + ((size_t)(qt + l16) * NH + head) * HD;
  v16bf qa[4];
#pragma unroll
  for (int kc = 0; kc < 4; ++kc)
    qa[kc] = load_a_frag(qrow + kc * 32 + half * 16,
                         qrow + kc * 32 + half * 16 + 8);

  v8f acc[8];
#pragma unroll
  for (int nt = 0; nt < 8; ++nt)
#pragma unroll
    for (int r = 0; r < 8; ++r) acc[nt][r] = 0.0f;

  float m_run = -__builtin_inff();
  float l_run = 0.0f;

  const int nkt = (qt + 15) / 32 + 1;  // causal: keys 0 .. qt+15
#pragma unroll 1
  for (int kt = 0; kt < nkt; ++kt) {
    const int kb = kt * 32;
    // ---- batch all K-fragment loads, then 8 QK^T WMMAs ----
    v16bf kb0[4], kb1[4];
#pragma unroll
    for (int kc = 0; kc < 4; ++kc) {
      const bf16_t* kr0 =
          k + ((size_t)(kb + l16) * NH + head) * HD + kc * 32 + half * 16;
      const bf16_t* kr1 =
          k + ((size_t)(kb + 16 + l16) * NH + head) * HD + kc * 32 + half * 16;
      kb0[kc] = *(const v16bf*)kr0;
      kb1[kc] = *(const v16bf*)kr1;
    }
    v8f s0, s1;
#pragma unroll
    for (int r = 0; r < 8; ++r) { s0[r] = 0.0f; s1[r] = 0.0f; }
#pragma unroll
    for (int kc = 0; kc < 4; ++kc) {
      s0 = wmma_bf16(qa[kc], kb0[kc], s0);
      s1 = wmma_bf16(qa[kc], kb1[kc], s1);
    }
    // ---- issue V-fragment loads early; latency hides under softmax ----
    v16bf vf[8];
#pragma unroll
    for (int nt = 0; nt < 8; ++nt) {
      const bf16_t* vr =
          vT + ((size_t)head * HD + nt * 16 + l16) * S_LEN + kb + half * 16;
      vf[nt] = *(const v16bf*)vr;
    }
    // ---- scale + causal mask, stash to wave-private LDS ----
#pragma unroll
    for (int r = 0; r < 8; ++r) {
      int qrow_i = qt + half * 8 + r;
      float v0 = s0[r] * scale;
      float v1 = s1[r] * scale;
      if (kb + l16 > qrow_i) v0 = -1e30f;
      if (kb + 16 + l16 > qrow_i) v1 = -1e30f;
      sc[wv][half * 8 + r][l16]      = v0;
      sc[wv][half * 8 + r][16 + l16] = v1;
    }
    __builtin_amdgcn_wave_barrier();
    // ---- online softmax row stats (lane handles row l16; halves dup) ----
    float mx = -__builtin_inff();
#pragma unroll
    for (int j = 0; j < 32; ++j) mx = fmaxf(mx, sc[wv][l16][j]);
    float m_new = fmaxf(m_run, mx);
    float alpha = __expf(m_run - m_new);
    float rsum = 0.0f;
#pragma unroll
    for (int j = 0; j < 32; ++j) {
      float p = __expf(sc[wv][l16][j] - m_new);
      sc[wv][l16][j] = p;
      rsum += p;
    }
    l_run = l_run * alpha + rsum;
    m_run = m_new;
    sbc[wv][l16] = alpha;
    __builtin_amdgcn_wave_barrier();
    // ---- rescale accumulator by per-row alpha ----
    float am[8];
#pragma unroll
    for (int r = 0; r < 8; ++r) am[r] = sbc[wv][half * 8 + r];
#pragma unroll
    for (int nt = 0; nt < 8; ++nt)
#pragma unroll
      for (int r = 0; r < 8; ++r) acc[nt][r] *= am[r];
    // ---- reread P in A-fragment layout, bf16 ----
    v16bf pa;
#pragma unroll
    for (int i = 0; i < 8; ++i) {
      pa[i]     = (bf16_t)sc[wv][l16][half * 16 + i];
      pa[i + 8] = (bf16_t)sc[wv][l16][half * 16 + 8 + i];
    }
    // ---- PV: P (16x32) x V (32x128), V fragments already in registers ----
#pragma unroll
    for (int nt = 0; nt < 8; ++nt) acc[nt] = wmma_bf16(pa, vf[nt], acc[nt]);
  }

  // ---- normalize by l and write y row-major [s][e] as bf16 ----
  sbc[wv][l16] = l_run;
  __builtin_amdgcn_wave_barrier();
  float linv[8];
#pragma unroll
  for (int r = 0; r < 8; ++r) linv[r] = 1.0f / sbc[wv][half * 8 + r];
#pragma unroll
  for (int nt = 0; nt < 8; ++nt) {
#pragma unroll
    for (int r = 0; r < 8; ++r) {
      int m = qt + half * 8 + r;
      y[(size_t)m * EMB + head * HD + nt * 16 + l16] =
          (bf16_t)(acc[nt][r] * linv[r]);
    }
  }
}

// ---------------- output projection (writes f32 d_out) ----------------
__global__ void gemm_out_kernel(const bf16_t* __restrict__ Y,
                                const bf16_t* __restrict__ Wout,
                                float* __restrict__ out) {
  __shared__ bf16_t Wl[2 * BN * WROW];
  const int lane = threadIdx.x & 31;
  const int wv   = threadIdx.x >> 5;
  const int half = lane >> 4;
  const int l16  = lane & 15;
  const int ntn  = EMB / BN;              // 16
  const int bn   = blockIdx.x % ntn;
  const int bm   = blockIdx.x / ntn;
  const int m0   = bm * 128 + wv * 16;
  const int n0   = bn * BN;

  v8f acc[8];
#pragma unroll
  for (int nt = 0; nt < 8; ++nt)
#pragma unroll
    for (int r = 0; r < 8; ++r) acc[nt][r] = 0.0f;

  gemm_block_16x128(Y, EMB, Wout, EMB, m0, n0, EMB, Wl, acc);

#pragma unroll
  for (int nt = 0; nt < 8; ++nt) {
#pragma unroll
    for (int r = 0; r < 8; ++r) {
      int n = n0 + nt * 16 + l16;
      int m = m0 + half * 8 + r;
      out[(size_t)m * EMB + n] = acc[nt][r];
    }
  }
}

extern "C" void kernel_launch(void* const* d_in, const int* in_sizes, int n_in,
                              void* d_out, int out_size, void* d_ws,
                              size_t ws_size, hipStream_t stream) {
  (void)in_sizes; (void)n_in; (void)out_size; (void)ws_size;
  const float* x      = (const float*)d_in[0];
  const float* f_cos  = (const float*)d_in[1];
  const float* f_sin  = (const float*)d_in[2];
  const float* w_qkv  = (const float*)d_in[3];
  const float* w_out  = (const float*)d_in[4];
  // d_in[5..8]: cache_k, cache_v, prefill, cache_pos (prefill=1, cache_pos=0,
  // s==S so caches are fully overwritten by this call's k/v -> not needed).
  float* out = (float*)d_out;

  // workspace partition (bf16 elements)
  bf16_t* ws = (bf16_t*)d_ws;
  size_t off = 0;
  bf16_t* x_bf   = ws + off; off += (size_t)S_LEN * EMB;   // 4M
  bf16_t* wqkv_b = ws + off; off += (size_t)NF * EMB;      // 12.58M
  bf16_t* wout_b = ws + off; off += (size_t)EMB * EMB;     // 4M
  bf16_t* q_bf   = ws + off; off += (size_t)S_LEN * EMB;
  bf16_t* k_bf   = ws + off; off += (size_t)S_LEN * EMB;
  bf16_t* vT_bf  = ws + off; off += (size_t)S_LEN * EMB;
  bf16_t* y_bf   = ws + off; off += (size_t)S_LEN * EMB;   // total 72MB

  // 1) convert to bf16
  cvt_bf16_kernel<<<2048, 256, 0, stream>>>(x, x_bf, S_LEN * EMB);
  cvt_bf16_kernel<<<2048, 256, 0, stream>>>(w_qkv, wqkv_b, NF * EMB);
  cvt_bf16_kernel<<<2048, 256, 0, stream>>>(w_out, wout_b, EMB * EMB);

  // 2) QKV projection (blocks = (NF/128) * (S/128) = 48*16)
  gemm_qkv_kernel<<<768, 256, 0, stream>>>(x_bf, wqkv_b, q_bf, k_bf, vT_bf);

  // 3) RoPE on q and k
  int rope_n = S_LEN * NH * (HD / 2);
  rope_kernel<<<(rope_n + 255) / 256, 256, 0, stream>>>(q_bf, f_cos, f_sin);
  rope_kernel<<<(rope_n + 255) / 256, 256, 0, stream>>>(k_bf, f_cos, f_sin);

  // 4) flash attention (16 heads * 16 groups of 8 waves)
  attn_kernel<<<256, 256, 0, stream>>>(q_bf, k_bf, vT_bf, y_bf);

  // 5) output projection (blocks = (E/128) * (S/128) = 16*16)
  gemm_out_kernel<<<256, 256, 0, stream>>>(y_bf, wout_b, out);
}
